// TriangleMultiplicationUpdate_14035953123352
// MI455X (gfx1250) — matrix operand order repaired and checked by reading, not compiled, not run on hardware
//
#include <hip/hip_runtime.h>
#include <hip/hip_bf16.h>

// ---------------------------------------------------------------------------
// TriangleMultiplicationUpdate (outgoing) for MI455X / gfx1250, wave32 WMMA.
// GEMMs via v_wmma_f32_16x16x32_bf16; block-shared operands staged into LDS
// with GLOBAL_LOAD_ASYNC_TO_LDS_B128 (ASYNCcnt + s_wait_asynccnt + barriers).
// Workspace carve: x_bf | a_ws[c][i][k] | b_ws[c][j][k] | g_ws[m][c] |
//                  upd_ws[m][c] (64 MiB each) | wbf (6*32 KiB bf16 weights).
// ---------------------------------------------------------------------------

typedef __bf16 bf16;
typedef __attribute__((ext_vector_type(16))) __bf16 v16bf;
typedef __attribute__((ext_vector_type(8)))  __bf16 v8bf;
typedef __attribute__((ext_vector_type(4)))  __bf16 v4bf;
typedef __attribute__((ext_vector_type(8)))  float  v8f;
typedef __attribute__((ext_vector_type(4)))  float  v4f;

#define LN_EPS 1e-5f
#define NN 512
#define CC 128
#define M_TOT (NN * NN)   // 262144 flattened rows
#define WMAT 16384        // 128*128 elements per weight matrix
#define KC 64             // staged K-chunk for the contraction
#define APAD 72           // padded LDS row stride (elems) for 64-wide chunks
#define WLD 136           // padded LDS row stride (elems) for 128-wide weights

static __device__ __forceinline__ bf16 f2bf(float f) {
  unsigned u = __float_as_uint(f);
  u += 0x7FFFu + ((u >> 16) & 1u);  // round-to-nearest-even
  return __builtin_bit_cast(bf16, (unsigned short)(u >> 16));
}
static __device__ __forceinline__ float bf2f(bf16 h) {
  unsigned short s = __builtin_bit_cast(unsigned short, h);
  return __uint_as_float(((unsigned)s) << 16);
}
static __device__ __forceinline__ float sigmoidf(float x) {
  return 1.0f / (1.0f + __expf(-x));
}

// Generic pointer (known to be LDS) -> 32-bit LDS byte offset for async dst.
typedef __attribute__((address_space(3))) char lds_char;
static __device__ __forceinline__ unsigned lds_off(const void* p) {
  return (unsigned)(uintptr_t)(lds_char*)p;
}
// Async copy 16B global -> LDS; tracked by ASYNCcnt.
static __device__ __forceinline__ void async_b128(unsigned ldst, const void* gsrc) {
  asm volatile("global_load_async_to_lds_b128 %0, %1, off"
               :: "v"(ldst), "v"(gsrc) : "memory");
}
#define S_WAIT_ASYNC(n) asm volatile("s_wait_asynccnt " #n ::: "memory")

// A-operand 16x32 fragment (ISA 05_wmma): lanes 0-15 row m hold K {0..7,16..23},
// lanes 16-31 same rows hold K {8..15,24..31}. Two 16B loads.
static __device__ __forceinline__ v16bf load_a_frag(const bf16* base, int ld,
                                                    int row0, int k0, int lane) {
  int m = lane & 15, hi = lane >> 4;
  const bf16* p = base + (size_t)(row0 + m) * ld + k0 + hi * 8;
  union { v16bf v; v8bf h[2]; } u;
  u.h[0] = *(const v8bf*)(p);
  u.h[1] = *(const v8bf*)(p + 16);
  return u.v;
}
// B-operand 32x16 fragment: lane n holds column n, K {hi*8+0..7, 16+hi*8+0..7}.
// Memory: base[col][k] row-major. Two 16B loads (padded-LDS safe).
static __device__ __forceinline__ v16bf load_b_frag(const bf16* base, int ld,
                                                    int col0, int k0, int lane) {
  int n = lane & 15, hi = lane >> 4;
  const bf16* p = base + (size_t)(col0 + n) * ld + k0 + hi * 8;
  union { v16bf v; v8bf h[2]; } u;
  u.h[0] = *(const v8bf*)(p);
  u.h[1] = *(const v8bf*)(p + 16);
  return u.v;
}

#define WMMA_BF16(a, b, c) \
  __builtin_amdgcn_wmma_f32_16x16x32_bf16(false, (a), false, (b), (short)0, (c), false, false)

// ---------------------------------------------------------------------------
// Kernel 1: f32 -> bf16 weight conversion (six 128x128 matrices).
__global__ void k_cvt_weights(const float* __restrict__ w0, const float* __restrict__ w1,
                              const float* __restrict__ w2, const float* __restrict__ w3,
                              const float* __restrict__ w4, const float* __restrict__ w5,
                              bf16* __restrict__ dst) {
  const float* srcs[6] = {w0, w1, w2, w3, w4, w5};
  const float* s = srcs[blockIdx.x];
  bf16* d = dst + (size_t)blockIdx.x * WMAT;
  for (int i = threadIdx.x; i < WMAT; i += blockDim.x) d[i] = f2bf(s[i]);
}

// ---------------------------------------------------------------------------
// Kernel 2: LayerNorm over C, f32 -> bf16. One wave per row.
__global__ void k_ln_pair(const float* __restrict__ x, const float* __restrict__ g,
                          const float* __restrict__ b, bf16* __restrict__ y) {
  int wave = threadIdx.x >> 5, lane = threadIdx.x & 31;
  size_t row = (size_t)blockIdx.x * 8 + wave;
  v4f v = *(const v4f*)(x + row * CC + lane * 4);
  float s  = v[0] + v[1] + v[2] + v[3];
  float s2 = v[0]*v[0] + v[1]*v[1] + v[2]*v[2] + v[3]*v[3];
  #pragma unroll
  for (int m = 16; m >= 1; m >>= 1) {
    s  += __shfl_xor(s,  m, 32);
    s2 += __shfl_xor(s2, m, 32);
  }
  float mu   = s * (1.0f / CC);
  float rstd = rsqrtf(s2 * (1.0f / CC) - mu * mu + LN_EPS);
  v4f gv = *(const v4f*)(g + lane * 4);
  v4f bv = *(const v4f*)(b + lane * 4);
  v4bf o;
  #pragma unroll
  for (int t = 0; t < 4; ++t) o[t] = f2bf((v[t] - mu) * rstd * gv[t] + bv[t]);
  *(v4bf*)(y + row * CC + lane * 4) = o;
}

// ---------------------------------------------------------------------------
// Kernel 3: five fused projections + sigmoid gating. 5 weight matrices are
// async-staged into LDS once per block (vs. 8x per-wave global re-reads).
__global__ void k_proj(const bf16* __restrict__ x, const bf16* __restrict__ wbf,
                       const float* __restrict__ ba1, const float* __restrict__ ba2,
                       const float* __restrict__ bb1, const float* __restrict__ bb2,
                       const float* __restrict__ bg,
                       bf16* __restrict__ a_ws, bf16* __restrict__ b_ws,
                       bf16* __restrict__ g_ws) {
  __shared__ bf16 Wlds[5 * CC * WLD];  // 5 matrices, padded rows (~170 KiB)
  int tid = threadIdx.x, wave = tid >> 5, lane = tid & 31;
  // Stage: 5*128 rows * 16 chunks(8 elems) = 10240 chunks, 40 per thread.
  #pragma unroll 1
  for (int it = 0; it < 40; ++it) {
    int ch  = tid + it * 256;
    int row = ch >> 4, c8 = ch & 15;               // row in 0..639
    unsigned dst = lds_off(&Wlds[row * WLD + c8 * 8]);
    async_b128(dst, wbf + (size_t)row * CC + c8 * 8);
  }
  S_WAIT_ASYNC(0);
  __syncthreads();

  int M0 = (blockIdx.x * 8 + wave) * 16;
  int n = lane & 15, hi = lane >> 4;
  v16bf A[4];
  #pragma unroll
  for (int t = 0; t < 4; ++t) A[t] = load_a_frag(x, CC, M0, t * 32, lane);

  const bf16* W0 = Wlds;
  const bf16* W1 = Wlds + 1 * CC * WLD;
  const bf16* W2 = Wlds + 2 * CC * WLD;
  const bf16* W3 = Wlds + 3 * CC * WLD;
  const bf16* W4 = Wlds + 4 * CC * WLD;

  #pragma unroll 1
  for (int nt = 0; nt < 8; ++nt) {
    int d0 = nt * 16;
    v8f acc0 = {}, acc1 = {}, acc2 = {}, acc3 = {}, acc4 = {};
    #pragma unroll
    for (int kt = 0; kt < 4; ++kt) {
      int k0 = kt * 32;
      acc0 = WMMA_BF16(A[kt], load_b_frag(W0, WLD, d0, k0, lane), acc0);
      acc1 = WMMA_BF16(A[kt], load_b_frag(W1, WLD, d0, k0, lane), acc1);
      acc2 = WMMA_BF16(A[kt], load_b_frag(W2, WLD, d0, k0, lane), acc2);
      acc3 = WMMA_BF16(A[kt], load_b_frag(W3, WLD, d0, k0, lane), acc3);
      acc4 = WMMA_BF16(A[kt], load_b_frag(W4, WLD, d0, k0, lane), acc4);
    }
    int c = d0 + n;
    float xa1 = ba1[c], xa2 = ba2[c], xb1 = bb1[c], xb2 = bb2[c], xg = bg[c];
    v8bf pa, pb;
    #pragma unroll
    for (int r = 0; r < 8; ++r) {   // VGPR r = row M0 + r + 8*hi, col d0 + n
      pa[r] = f2bf(sigmoidf(acc0[r] + xa1) * (acc1[r] + xa2));
      pb[r] = f2bf(sigmoidf(acc2[r] + xb1) * (acc3[r] + xb2));
      g_ws[(size_t)(M0 + r + 8 * hi) * CC + c] = f2bf(sigmoidf(acc4[r] + xg));
    }
    *(v8bf*)(a_ws + (size_t)c * M_TOT + M0 + 8 * hi) = pa;   // packed 16B
    *(v8bf*)(b_ws + (size_t)c * M_TOT + M0 + 8 * hi) = pb;
  }
}

// ---------------------------------------------------------------------------
// Kernel 4: triangle contraction, update_c = A_c * B_c^T per channel.
// Block = 8 waves -> 128x128 tile (waves 4x2, each 32x64). K chunks of 64 are
// double-buffered in LDS via async loads; per-block operand traffic from L2 is
// (128+128)*512*2B = 256 KiB (slabs themselves stay L2-resident: 192 MB L2).
__global__ void k_tri(const bf16* __restrict__ a_ws, const bf16* __restrict__ b_ws,
                      bf16* __restrict__ upd) {
  __shared__ bf16 Abuf[2][128 * APAD];   // 36 KiB
  __shared__ bf16 Bbuf[2][128 * APAD];   // 36 KiB
  int tid = threadIdx.x, wave = tid >> 5, lane = tid & 31;
  int c    = blockIdx.x;
  int Mblk = blockIdx.y * 128;
  int Nblk = blockIdx.z * 128;
  int n = lane & 15, hi = lane >> 4;
  const bf16* Ac = a_ws + (size_t)c * M_TOT;
  const bf16* Bc = b_ws + (size_t)c * M_TOT;

  // Stage one K-chunk (128 rows x 64 k) of A and B: 4+4 async b128 per thread.
  auto stage = [&](int buf, int kc0) {
    #pragma unroll
    for (int it = 0; it < 4; ++it) {
      int idx = (tid + it * 256) * 8;          // element index in chunk
      int row = idx >> 6, col = idx & 63;      // KC = 64
      unsigned loff = (unsigned)((row * APAD + col) * 2);
      async_b128(lds_off(&Abuf[buf][0]) + loff,
                 Ac + (size_t)(Mblk + row) * NN + kc0 + col);
      async_b128(lds_off(&Bbuf[buf][0]) + loff,
                 Bc + (size_t)(Nblk + row) * NN + kc0 + col);
    }
  };

  int wave_m = wave & 3, wave_n = wave >> 2;
  int M0w = wave_m * 32, N0w = wave_n * 64;

  v8f acc[2][4] = {};
  stage(0, 0);
  #pragma unroll 1
  for (int kc = 0; kc < NN / KC; ++kc) {
    int buf = kc & 1;
    if (kc + 1 < NN / KC) {
      stage(buf ^ 1, (kc + 1) * KC);
      S_WAIT_ASYNC(8);          // previous chunk's 8 loads complete
    } else {
      S_WAIT_ASYNC(0);
    }
    __syncthreads();
    const bf16* Ab = &Abuf[buf][0];
    const bf16* Bb = &Bbuf[buf][0];
    #pragma unroll
    for (int ks = 0; ks < 2; ++ks) {
      int k0 = ks * 32;
      v16bf a0 = load_a_frag(Ab, APAD, M0w,      k0, lane);
      v16bf a1 = load_a_frag(Ab, APAD, M0w + 16, k0, lane);
      #pragma unroll
      for (int nt = 0; nt < 4; ++nt) {
        v16bf bf_ = load_b_frag(Bb, APAD, N0w + nt * 16, k0, lane);
        acc[0][nt] = WMMA_BF16(a0, bf_, acc[0][nt]);
        acc[1][nt] = WMMA_BF16(a1, bf_, acc[1][nt]);
      }
    }
    __syncthreads();
  }

  #pragma unroll
  for (int mt = 0; mt < 2; ++mt)
    #pragma unroll
    for (int nt = 0; nt < 4; ++nt)
      #pragma unroll
      for (int r = 0; r < 8; ++r) {
        int i = Mblk + M0w + mt * 16 + r + 8 * hi;
        int j = Nblk + N0w + nt * 16 + n;
        upd[((size_t)i * NN + j) * CC + c] = f2bf(acc[mt][nt][r]);
      }
}

// ---------------------------------------------------------------------------
// Kernel 5: LayerNorm over T (per-row via shfl) + final projection with w_u
// (async-staged to LDS), gated by sigmoid'd g. f32 output.
__global__ void k_out(const bf16* __restrict__ upd, const bf16* __restrict__ wu,
                      const float* __restrict__ lng, const float* __restrict__ lnb,
                      const float* __restrict__ bu, const bf16* __restrict__ g_ws,
                      float* __restrict__ out) {
  __shared__ bf16 Wlds[CC * WLD];   // ~34 KiB
  int tid = threadIdx.x, wave = tid >> 5, lane = tid & 31;
  #pragma unroll 1
  for (int it = 0; it < 8; ++it) {  // 128 rows * 16 chunks / 256 threads
    int ch  = tid + it * 256;
    int row = ch >> 4, c8 = ch & 15;
    async_b128(lds_off(&Wlds[row * WLD + c8 * 8]), wu + (size_t)row * CC + c8 * 8);
  }
  S_WAIT_ASYNC(0);
  __syncthreads();

  int M0 = (blockIdx.x * 8 + wave) * 16;
  int n = lane & 15, hi = lane >> 4;

  v16bf U[4];
  float s = 0.f, s2 = 0.f;
  #pragma unroll
  for (int t = 0; t < 4; ++t) {
    U[t] = load_a_frag(upd, CC, M0, t * 32, lane);
    #pragma unroll
    for (int e = 0; e < 16; ++e) { float f = bf2f(U[t][e]); s += f; s2 += f * f; }
  }
  s  += __shfl_xor(s,  16, 32);   // row split between lane L and L+16
  s2 += __shfl_xor(s2, 16, 32);
  float mu   = s * (1.0f / CC);
  float rstd = rsqrtf(s2 * (1.0f / CC) - mu * mu + LN_EPS);
  #pragma unroll
  for (int t = 0; t < 4; ++t)
    #pragma unroll
    for (int e = 0; e < 16; ++e) {
      int k = t * 32 + hi * 8 + (e & 7) + ((e >> 3) * 16);  // A-frag channel map
      U[t][e] = f2bf((bf2f(U[t][e]) - mu) * rstd * lng[k] + lnb[k]);
    }

  #pragma unroll 1
  for (int nt = 0; nt < 8; ++nt) {
    int d0 = nt * 16;
    v8f acc = {};
    #pragma unroll
    for (int kt = 0; kt < 4; ++kt)
      acc = WMMA_BF16(U[kt], load_b_frag(Wlds, WLD, d0, kt * 32, lane), acc);
    int d = d0 + n;
    float bias = bu[d];
    #pragma unroll
    for (int r = 0; r < 8; ++r) {
      size_t m = (size_t)(M0 + r + 8 * hi);
      out[m * CC + d] = (acc[r] + bias) * bf2f(g_ws[m * CC + d]);
    }
  }
}

// ---------------------------------------------------------------------------
extern "C" void kernel_launch(void* const* d_in, const int* in_sizes, int n_in,
                              void* d_out, int out_size, void* d_ws, size_t ws_size,
                              hipStream_t stream) {
  const float* pair  = (const float*)d_in[0];
  const float* lnpg  = (const float*)d_in[1];
  const float* lnpb  = (const float*)d_in[2];
  const float* w_a1  = (const float*)d_in[3];
  const float* b_a1  = (const float*)d_in[4];
  const float* w_a2  = (const float*)d_in[5];
  const float* b_a2  = (const float*)d_in[6];
  const float* w_b1  = (const float*)d_in[7];
  const float* b_b1  = (const float*)d_in[8];
  const float* w_b2  = (const float*)d_in[9];
  const float* b_b2  = (const float*)d_in[10];
  const float* w_g   = (const float*)d_in[11];
  const float* b_g   = (const float*)d_in[12];
  const float* ln_ug = (const float*)d_in[13];
  const float* ln_ub = (const float*)d_in[14];
  const float* w_u   = (const float*)d_in[15];
  const float* b_u   = (const float*)d_in[16];

  const size_t SLAB = (size_t)M_TOT * CC * sizeof(bf16);  // 64 MiB
  char* ws = (char*)d_ws;
  bf16* x_bf   = (bf16*)(ws + 0 * SLAB);
  bf16* a_ws   = (bf16*)(ws + 1 * SLAB);
  bf16* b_ws   = (bf16*)(ws + 2 * SLAB);
  bf16* g_ws   = (bf16*)(ws + 3 * SLAB);
  bf16* upd_ws = (bf16*)(ws + 4 * SLAB);
  bf16* wbf    = (bf16*)(ws + 5 * SLAB);

  k_cvt_weights<<<6, 256, 0, stream>>>(w_a1, w_a2, w_b1, w_b2, w_g, w_u, wbf);
  k_ln_pair<<<M_TOT / 8, 256, 0, stream>>>(pair, lnpg, lnpb, x_bf);
  k_proj<<<M_TOT / 16 / 8, 256, 0, stream>>>(x_bf, wbf, b_a1, b_a2, b_b1, b_b2,
                                             b_g, a_ws, b_ws, g_ws);
  k_tri<<<dim3(CC, NN / 128, NN / 128), 256, 0, stream>>>(a_ws, b_ws, upd_ws);
  k_out<<<M_TOT / 16 / 8, 256, 0, stream>>>(upd_ws, wbf + 5 * WMAT, ln_ug, ln_ub,
                                            b_u, g_ws, (float*)d_out);
}
